// EfficientDeformableAttention_ori_17703855194637
// MI455X (gfx1250) — matrix operand order repaired and checked by reading, not compiled, not run on hardware
//
#include <hip/hip_runtime.h>
#include <stdint.h>
#include <stddef.h>

// Problem constants (reference shapes)
constexpr int kB     = 4;
constexpr int kNq    = 8192;     // log2 = 13
constexpr int kC     = 256;
constexpr int kH     = 128;
constexpr int kW     = 128;
constexpr int kHW    = kH * kW;  // 16384, log2 = 14
constexpr int kHeads = 8;
constexpr int kP     = 4;
constexpr int kHd    = 32;

typedef __attribute__((ext_vector_type(16))) __bf16 v16bf;
typedef __attribute__((ext_vector_type(8)))  float  v8f;

union Frag {
    v16bf v;
    uint4 q[2];
    __bf16 h[16];
    unsigned short s[16];
};

// Hardware f32 <-> bf16 conversion (lowers to v_cvt_pk_bf16_f32-class ops)
__device__ __forceinline__ unsigned short f2bf(float f) {
    __bf16 b = (__bf16)f;
    return __builtin_bit_cast(unsigned short, b);
}
__device__ __forceinline__ float bf2f(unsigned short s) {
    return (float)__builtin_bit_cast(__bf16, s);
}

// ---------------------------------------------------------------------------
// Prep: transpose f32 [K][N] row-major -> bf16 [N][K] (K contiguous per col)
// ---------------------------------------------------------------------------
__global__ void transpose_convert(const float* __restrict__ src,
                                  unsigned short* __restrict__ dst,
                                  int K, int N) {
    int idx = blockIdx.x * blockDim.x + threadIdx.x;
    if (idx >= K * N) return;
    int k = idx / N;
    int n = idx - k * N;
    dst[(size_t)n * K + k] = f2bf(src[idx]);
}

// Build concatenated [W_off | W_attn] transposed bf16 [96][256] and bias[96]
__global__ void build_cat(const float* __restrict__ Woff,
                          const float* __restrict__ boff,
                          const float* __restrict__ Wattn,
                          const float* __restrict__ battn,
                          unsigned short* __restrict__ WcatT,
                          float* __restrict__ bcat) {
    int idx = blockIdx.x * blockDim.x + threadIdx.x;
    const int total = 96 * kC;
    if (idx < total) {
        int n = idx / kC;
        int k = idx - n * kC;
        float v = (n < 64) ? Woff[(size_t)k * 64 + n]
                           : Wattn[(size_t)k * 32 + (n - 64)];
        WcatT[(size_t)n * kC + k] = f2bf(v);
    } else if (idx < total + 96) {
        int j = idx - total;
        bcat[j] = (j < 64) ? boff[j] : battn[j - 64];
    }
}

// ---------------------------------------------------------------------------
// WMMA GEMM: Out[M,N] = A[M,K] * Bt^T + bias   (Bt is bf16 [N][K])
// Wave tile: 16 x (16*NT). AT = float (convert) or ushort (bf16 raw).
// PERM: scatter output into [b][head][pix][hd] layout (value projection).
// ---------------------------------------------------------------------------
template <int NT, typename AT, typename OT, bool PERM>
__global__ __launch_bounds__(128) void gemm_bias_wmma(
    const AT* __restrict__ A, const unsigned short* __restrict__ Bt,
    const float* __restrict__ bias, OT* __restrict__ Out,
    int M, int N, int K) {
    const int lane = threadIdx.x & 31;
    const int wave = threadIdx.x >> 5;
    const int tile = blockIdx.x * 4 + wave;
    const int tilesN = N / (16 * NT);
    const int tm = tile / tilesN;
    const int tn = tile - tm * tilesN;

    const int mrow  = (tm << 4) + (lane & 15);        // A row for this lane
    const int koff  = (lane & 16) ? 8 : 0;            // A K sub-offset (ISA layout)
    const int ncol  = tn * (16 * NT) + (lane & 15);   // first B column
    const int kboff = (lane & 16) ? 8 : 0;            // B K sub-offset (ISA layout)

    v8f acc[NT];
#pragma unroll
    for (int t = 0; t < NT; ++t) acc[t] = v8f{};

    for (int kk = 0; kk < K; kk += 32) {
        Frag a;
        if constexpr (sizeof(AT) == 4) {
            const float* ap = (const float*)A + (size_t)mrow * K + kk + koff;
            float4 f0 = ((const float4*)ap)[0];
            float4 f1 = ((const float4*)ap)[1];
            float4 f2 = ((const float4*)(ap + 16))[0];
            float4 f3 = ((const float4*)(ap + 16))[1];
            a.h[0]  = (__bf16)f0.x; a.h[1]  = (__bf16)f0.y;
            a.h[2]  = (__bf16)f0.z; a.h[3]  = (__bf16)f0.w;
            a.h[4]  = (__bf16)f1.x; a.h[5]  = (__bf16)f1.y;
            a.h[6]  = (__bf16)f1.z; a.h[7]  = (__bf16)f1.w;
            a.h[8]  = (__bf16)f2.x; a.h[9]  = (__bf16)f2.y;
            a.h[10] = (__bf16)f2.z; a.h[11] = (__bf16)f2.w;
            a.h[12] = (__bf16)f3.x; a.h[13] = (__bf16)f3.y;
            a.h[14] = (__bf16)f3.z; a.h[15] = (__bf16)f3.w;
        } else {
            const unsigned short* ap = (const unsigned short*)A + (size_t)mrow * K + kk + koff;
            a.q[0] = ((const uint4*)ap)[0];              // K .. K+7
            a.q[1] = *((const uint4*)(ap + 16));         // K+16 .. K+23
        }
#pragma unroll
        for (int t = 0; t < NT; ++t) {
            Frag b;
            const unsigned short* bp = Bt + (size_t)(ncol + 16 * t) * K + kk + kboff;
            b.q[0] = ((const uint4*)bp)[0];
            b.q[1] = *((const uint4*)(bp + 16));
            acc[t] = __builtin_amdgcn_wmma_f32_16x16x32_bf16(
                false, a.v, false, b.v, (short)0, acc[t], false, false);
        }
    }

    const int rbase = (tm << 4) + ((lane & 16) ? 8 : 0);
#pragma unroll
    for (int t = 0; t < NT; ++t) {
        const int c = ncol + 16 * t;
        const float bia = bias[c];
#pragma unroll
        for (int r = 0; r < 8; ++r) {
            const int row = rbase + r;
            const float val = acc[t][r] + bia;
            size_t o;
            if constexpr (PERM) {
                // row = b*HW + pix ; c = head*32 + d -> [b][head][pix][d]
                const int bIdx = row >> 14;              // HW = 16384
                const int pix  = row & (kHW - 1);
                o = (size_t)bIdx * (kHeads * kHW * kHd)
                  + (size_t)(c >> 5) * (kHW * kHd)
                  + (size_t)pix * kHd + (c & 31);
            } else {
                o = (size_t)row * N + c;
            }
            if constexpr (sizeof(OT) == 2) Out[o] = f2bf(val);
            else                           Out[o] = val;
        }
    }
}

// ---------------------------------------------------------------------------
// Sampling + per-head softmax + weighted aggregation.
// One wave per query; lane = hd channel. vbf layout [b][head][pix][hd] bf16.
// ---------------------------------------------------------------------------
__global__ __launch_bounds__(256) void sample_agg(
    const float* __restrict__ refpts,          // [B*Nq][2]
    const float* __restrict__ qproj,           // [B*Nq][96] = off(64)|attn(32)
    const unsigned short* __restrict__ vbf,    // [B][heads][HW][hd] bf16
    unsigned short* __restrict__ aggbf) {      // [B*Nq][256] bf16
    const int lane = threadIdx.x & 31;
    const int q = blockIdx.x * 8 + (threadIdx.x >> 5);
    const int b = q >> 13;                     // Nq = 8192
    const int d = lane;

    const float refx = refpts[(size_t)q * 2 + 0];
    const float refy = refpts[(size_t)q * 2 + 1];
    const float* qp = qproj + (size_t)q * 96;

    float acc[kHeads];
#pragma unroll
    for (int h = 0; h < kHeads; ++h) acc[h] = 0.0f;

    for (int h = 0; h < kHeads; ++h) {
        // softmax over the 4 points
        float l0 = qp[64 + h * 4 + 0], l1 = qp[64 + h * 4 + 1];
        float l2 = qp[64 + h * 4 + 2], l3 = qp[64 + h * 4 + 3];
        float m = fmaxf(fmaxf(l0, l1), fmaxf(l2, l3));
        float e[4];
        e[0] = __expf(l0 - m); e[1] = __expf(l1 - m);
        e[2] = __expf(l2 - m); e[3] = __expf(l3 - m);
        float inv = 1.0f / (e[0] + e[1] + e[2] + e[3]);

        const unsigned short* vb = vbf + (size_t)(b * kHeads + h) * kHW * kHd;

#pragma unroll
        for (int p = 0; p < kP; ++p) {
            const float wp = e[p] * inv;
            const float ox = qp[h * 8 + p * 2 + 0] * (0.1f / (float)kW);
            const float oy = qp[h * 8 + p * 2 + 1] * (0.1f / (float)kH);
            const float lx = fminf(fmaxf(refx + ox, 0.0f), 1.0f);
            const float ly = fminf(fmaxf(refy + oy, 0.0f), 1.0f);
            const float x = lx * (float)kW - 0.5f;
            const float y = ly * (float)kH - 0.5f;
            const float xf = floorf(x), yf = floorf(y);
            const float wx = x - xf, wy = y - yf;
            const int ix0 = (int)xf, iy0 = (int)yf;
            const int ix1 = ix0 + 1, iy1 = iy0 + 1;

            const float cw[4] = { (1.0f - wx) * (1.0f - wy), wx * (1.0f - wy),
                                  (1.0f - wx) * wy,          wx * wy };
            const int cx[4] = { ix0, ix1, ix0, ix1 };
            const int cy[4] = { iy0, iy0, iy1, iy1 };
#pragma unroll
            for (int c = 0; c < 4; ++c) {
                if (cx[c] >= 0 && cx[c] < kW && cy[c] >= 0 && cy[c] < kH) {
                    float v = bf2f(vb[(size_t)(cy[c] * kW + cx[c]) * kHd + d]);
                    acc[h] += wp * cw[c] * v;
                }
            }
        }
    }

    unsigned short* out = aggbf + (size_t)q * kC;
#pragma unroll
    for (int h = 0; h < kHeads; ++h) out[h * kHd + d] = f2bf(acc[h]);
}

// ---------------------------------------------------------------------------
extern "C" void kernel_launch(void* const* d_in, const int* in_sizes, int n_in,
                              void* d_out, int out_size, void* d_ws, size_t ws_size,
                              hipStream_t stream) {
    const float* query  = (const float*)d_in[0];
    const float* refpts = (const float*)d_in[1];
    const float* value  = (const float*)d_in[2];
    const float* W_off  = (const float*)d_in[3];
    const float* b_off  = (const float*)d_in[4];
    const float* W_attn = (const float*)d_in[5];
    const float* b_attn = (const float*)d_in[6];
    const float* W_v    = (const float*)d_in[7];
    const float* b_v    = (const float*)d_in[8];
    const float* W_out  = (const float*)d_in[9];
    const float* b_out  = (const float*)d_in[10];
    float* out = (float*)d_out;

    // Workspace layout (bytes, all 256-aligned)
    char* ws = (char*)d_ws;
    unsigned short* vbf   = (unsigned short*)(ws);                       // 32 MB
    float*          qproj = (float*)(ws + 33554432);                     // 12 MB
    unsigned short* aggbf = (unsigned short*)(ws + 33554432 + 12582912); // 16 MB
    unsigned short* WvT   = (unsigned short*)(ws + 62914560);
    unsigned short* WoutT = (unsigned short*)(ws + 62914560 + 131072);
    unsigned short* WcatT = (unsigned short*)(ws + 62914560 + 262144);
    float*          bcat  = (float*)(ws + 62914560 + 262144 + 49152);

    // 1) weight prep
    transpose_convert<<<(kC * kC + 255) / 256, 256, 0, stream>>>(W_v, WvT, kC, kC);
    transpose_convert<<<(kC * kC + 255) / 256, 256, 0, stream>>>(W_out, WoutT, kC, kC);
    build_cat<<<(96 * kC + 96 + 255) / 256, 256, 0, stream>>>(W_off, b_off, W_attn,
                                                              b_attn, WcatT, bcat);

    // 2) value projection -> bf16 [b][head][pix][hd]; M=65536 N=256 K=256
    //    tiles = (65536/16)*(256/64) = 16384 -> 4096 blocks of 4 waves
    gemm_bias_wmma<4, float, unsigned short, true>
        <<<4096, 128, 0, stream>>>(value, WvT, b_v, vbf, kB * kHW, kC, kC);

    // 3) offsets+attn projection; M=32768 N=96 K=256
    //    tiles = (32768/16)*(96/32) = 6144 -> 1536 blocks
    gemm_bias_wmma<2, float, float, false>
        <<<1536, 128, 0, stream>>>(query, WcatT, bcat, qproj, kB * kNq, 96, kC);

    // 4) bilinear sampling + softmax-weighted aggregation (one wave per query)
    sample_agg<<<(kB * kNq) / 8, 256, 0, stream>>>(refpts, qproj, vbf, aggbf);

    // 5) output projection; M=32768 N=256 K=256
    //    tiles = (32768/16)*(256/64) = 8192 -> 2048 blocks
    gemm_bias_wmma<4, unsigned short, float, false>
        <<<2048, 128, 0, stream>>>(aggbf, WoutT, b_out, out, kB * kNq, kC, kC);

    (void)in_sizes; (void)n_in; (void)out_size; (void)ws_size;
}
